// Temporal_dot_80951543595344
// MI455X (gfx1250) — compile-verified, matrix operand order checked by reading, not run on hardware
//
#include <hip/hip_runtime.h>

// ---------------------------------------------------------------------------
// Types
// ---------------------------------------------------------------------------
typedef __attribute__((ext_vector_type(16))) __bf16 v16bf;
typedef __attribute__((ext_vector_type(8)))  float  v8f;
typedef __attribute__((ext_vector_type(4)))  unsigned int u32x4;
typedef __attribute__((ext_vector_type(4)))  float  f32x4;

union Frag {            // 16 bf16 = 32 bytes = 8 VGPRs (one WMMA A/B operand)
    u32x4 q[2];
    v16bf v;
};

__device__ __forceinline__ unsigned short f32_bf16(float f) {
    union { float f; unsigned u; } c; c.f = f;
    unsigned u = c.u;
    u += 0x7fffu + ((u >> 16) & 1u);   // round-to-nearest-even
    return (unsigned short)(u >> 16);
}

// ---------------------------------------------------------------------------
// Register staging helpers (double buffer: global -> regs -> LDS as bf16)
// ---------------------------------------------------------------------------
template <typename TIN> struct Stage16 {           // 16 elements per thread
    f32x4 f[4]; u32x4 u[2];
    __device__ __forceinline__ void load(const TIN* src) {
        if constexpr (sizeof(TIN) == 4) {
            f[0] = ((const f32x4*)src)[0]; f[1] = ((const f32x4*)src)[1];
            f[2] = ((const f32x4*)src)[2]; f[3] = ((const f32x4*)src)[3];
        } else {
            u[0] = ((const u32x4*)src)[0]; u[1] = ((const u32x4*)src)[1];
        }
    }
    __device__ __forceinline__ void store(unsigned short* dst) {
        if constexpr (sizeof(TIN) == 4) {
            union { unsigned short s[16]; u32x4 q[2]; } t;
#pragma unroll
            for (int v = 0; v < 4; ++v)
#pragma unroll
                for (int c = 0; c < 4; ++c) t.s[v * 4 + c] = f32_bf16(f[v][c]);
            ((u32x4*)dst)[0] = t.q[0]; ((u32x4*)dst)[1] = t.q[1];
        } else {
            ((u32x4*)dst)[0] = u[0]; ((u32x4*)dst)[1] = u[1];
        }
    }
};
template <typename TIN> struct Stage8 {            // 8 elements per thread
    f32x4 f[2]; u32x4 u;
    __device__ __forceinline__ void load(const TIN* src) {
        if constexpr (sizeof(TIN) == 4) {
            f[0] = ((const f32x4*)src)[0]; f[1] = ((const f32x4*)src)[1];
        } else {
            u = *(const u32x4*)src;
        }
    }
    __device__ __forceinline__ void store(unsigned short* dst) {
        if constexpr (sizeof(TIN) == 4) {
            union { unsigned short s[8]; u32x4 q; } t;
#pragma unroll
            for (int v = 0; v < 2; ++v)
#pragma unroll
                for (int c = 0; c < 4; ++c) t.s[v * 4 + c] = f32_bf16(f[v][c]);
            *(u32x4*)dst = t.q;
        } else {
            *(u32x4*)dst = u;
        }
    }
};

// ---------------------------------------------------------------------------
// Tiled WMMA GEMM:  C[M,N] = A[M,K] * W[N,K]^T (+ bias[N])
//   A, W row-major; TIN = float (f32->bf16 during staging) or ushort (bf16).
//   Block tile: 128(M) x NTILE(N), 256 threads = 8 waves (wave32), K-step 32.
//   Waves arranged 4(M) x 2(N); wave tile = 32 x (NTILE/2).
//   Register double-buffered staging; bank-conflict-free LDS (row = 40 elems).
//   STORE_MODE: 0 = f32 row-major (batched via blockIdx.z / strideC)
//               1 = bf16 row-major
//               2 = bf16 batch-transposed: C_t[b][n][t], b=m>>7, t=m&127
// ---------------------------------------------------------------------------
template <typename TIN, int NTILE, int STORE_MODE, bool HAS_BIAS>
__global__ void __launch_bounds__(256)
wmma_gemm(const TIN* __restrict__ A, const TIN* __restrict__ W,
          const float* __restrict__ bias, void* __restrict__ Cout,
          int K, int N,
          long strideA, long strideW, long strideC)
{
    constexpr int KT  = 32;
    constexpr int LDA = 40;                     // bf16 elems; 80B row, 16B aligned
    constexpr int NJ  = NTILE / 32;             // WMMA accum tiles per wave in N
    __shared__ __align__(16) unsigned short As[128 * LDA];
    __shared__ __align__(16) unsigned short Ws[NTILE * LDA];

    const int tid = threadIdx.x;
    const int bz  = blockIdx.z;
    A += (long)bz * strideA;
    W += (long)bz * strideW;

    const int mblock = blockIdx.y * 128;
    const int nblock = blockIdx.x * NTILE;

    const int wave = tid >> 5;
    const int lane = tid & 31;
    const int lr   = lane & 15;
    const int lh   = lane >> 4;
    const int wm   = (wave & 3) * 32;            // wave M offset in block tile
    const int wn   = (wave >> 2) * (NTILE / 2);  // wave N offset in block tile

    // staging work split
    const int a_row = tid >> 1;                  // 0..127
    const int a_col = (tid & 1) * 16;            // 0 or 16
    const int w_row = (NTILE == 128) ? (tid >> 1) : (tid >> 2);
    const int w_col = (NTILE == 128) ? ((tid & 1) * 16) : ((tid & 3) * 8);

    const TIN* Abase = A + (long)(mblock + a_row) * K + a_col;
    const TIN* Wbase = W + (long)(nblock + w_row) * K + w_col;

    Stage16<TIN> ra;
    Stage16<TIN> rw16;
    Stage8<TIN>  rw8;

    auto load_tiles = [&](int k0) {
        ra.load(Abase + k0);
        if constexpr (NTILE == 128) rw16.load(Wbase + k0);
        else                        rw8.load(Wbase + k0);
        if (k0 + KT < K) __builtin_prefetch(Abase + k0 + KT, 0, 1);
    };

    v8f acc[2][NJ] = {};

    load_tiles(0);
    for (int k0 = 0; k0 < K; k0 += KT) {
        // regs -> LDS (bf16)
        ra.store(&As[a_row * LDA + a_col]);
        if constexpr (NTILE == 128) rw16.store(&Ws[w_row * LDA + w_col]);
        else                        rw8.store(&Ws[w_row * LDA + w_col]);
        __syncthreads();

        // issue next tile's global loads while WMMAs run from LDS
        if (k0 + KT < K) load_tiles(k0 + KT);

        Frag af[2], bfr[NJ];
#pragma unroll
        for (int i = 0; i < 2; ++i) {
            const unsigned short* pa = &As[(wm + i * 16 + lr) * LDA + lh * 8];
            af[i].q[0] = *(const u32x4*)pa;
            af[i].q[1] = *(const u32x4*)(pa + 16);
        }
#pragma unroll
        for (int j = 0; j < NJ; ++j) {
            const unsigned short* pw = &Ws[(wn + j * 16 + lr) * LDA + lh * 8];
            bfr[j].q[0] = *(const u32x4*)pw;
            bfr[j].q[1] = *(const u32x4*)(pw + 16);
        }
#pragma unroll
        for (int i = 0; i < 2; ++i)
#pragma unroll
            for (int j = 0; j < NJ; ++j)
                acc[i][j] = __builtin_amdgcn_wmma_f32_16x16x32_bf16(
                    false, af[i].v, false, bfr[j].v, (short)0, acc[i][j], false, false);
        __syncthreads();
    }

    // ---- epilogue: bias + store ----
#pragma unroll
    for (int j = 0; j < NJ; ++j) {
        const int ng = nblock + wn + j * 16 + lr;      // lane owns one column
        float bv = 0.0f;
        if constexpr (HAS_BIAS) bv = bias[ng];
#pragma unroll
        for (int i = 0; i < 2; ++i) {
            const int mg0 = mblock + wm + i * 16 + lh * 8;
#pragma unroll
            for (int r = 0; r < 8; ++r) {
                const float v = acc[i][j][r] + bv;
                const long  mg = mg0 + r;
                if constexpr (STORE_MODE == 0) {
                    ((float*)Cout)[(long)bz * strideC + mg * (long)N + ng] = v;
                } else if constexpr (STORE_MODE == 1) {
                    ((unsigned short*)Cout)[mg * (long)N + ng] = f32_bf16(v);
                } else {
                    const long bb = mg >> 7, tt = mg & 127;   // batch-transposed
                    ((unsigned short*)Cout)[(bb * (long)N + ng) * 128 + tt] = f32_bf16(v);
                }
            }
        }
    }
}

// ---------------------------------------------------------------------------
// Bulk f32 -> bf16 conversion (8 elems/thread, b128 in / b128 out)
// ---------------------------------------------------------------------------
__global__ void __launch_bounds__(256)
cvt_f32_bf16(const float* __restrict__ src, unsigned short* __restrict__ dst, long n)
{
    const long i = ((long)blockIdx.x * 256 + threadIdx.x) * 8;
    if (i >= n) return;
    f32x4 a = ((const f32x4*)(src + i))[0];
    f32x4 b = ((const f32x4*)(src + i))[1];
    union { unsigned short s[8]; u32x4 q; } t;
#pragma unroll
    for (int c = 0; c < 4; ++c) { t.s[c] = f32_bf16(a[c]); t.s[4 + c] = f32_bf16(b[c]); }
    *(u32x4*)(dst + i) = t.q;
}

// ---------------------------------------------------------------------------
// Fused scores + activation:
//   scores[b] = q[b](128x64) * k[b](128x64)^T  via WMMA bf16
//   act = w0*relu + w1*sigmoid + w2*softmax(scores, axis=t)  -> bf16 [b,t,s]
// ---------------------------------------------------------------------------
__global__ void __launch_bounds__(256)
scores_act(const unsigned short* __restrict__ q,
           const unsigned short* __restrict__ k,
           const float* __restrict__ aw,
           unsigned short* __restrict__ act)
{
    constexpr int T = 128, NF = 64, SLD = 132;       // padded f32 row
    __shared__ __align__(16) float sc[T * SLD];      // ~67.6 KB of 320 KB LDS

    const int b    = blockIdx.x;
    const int tid  = threadIdx.x;
    const int wave = tid >> 5;
    const int lane = tid & 31;
    const int lr   = lane & 15;
    const int lh   = lane >> 4;

    const unsigned short* qb = q + (long)b * T * NF;
    const unsigned short* kb = k + (long)b * T * NF;

    v8f acc[8] = {};
#pragma unroll
    for (int ks = 0; ks < 2; ++ks) {                 // K = 64 = 2 x 32
        Frag a;
        const unsigned short* pa = qb + (wave * 16 + lr) * NF + ks * 32 + lh * 8;
        a.q[0] = *(const u32x4*)pa;
        a.q[1] = *(const u32x4*)(pa + 16);
#pragma unroll
        for (int j = 0; j < 8; ++j) {
            Frag bb;
            const unsigned short* pb = kb + (j * 16 + lr) * NF + ks * 32 + lh * 8;
            bb.q[0] = *(const u32x4*)pb;
            bb.q[1] = *(const u32x4*)(pb + 16);
            acc[j] = __builtin_amdgcn_wmma_f32_16x16x32_bf16(
                false, a.v, false, bb.v, (short)0, acc[j], false, false);
        }
    }
#pragma unroll
    for (int j = 0; j < 8; ++j) {
        const int n = j * 16 + lr;
#pragma unroll
        for (int r = 0; r < 8; ++r) {
            const int m = wave * 16 + lh * 8 + r;
            sc[m * SLD + n] = acc[j][r];
        }
    }
    __syncthreads();

    // column-wise softmax over t (reference axis=1) + fused activations
    if (tid < T) {
        const int s = tid;
        const float w0 = aw[0], w1 = aw[1], w2 = aw[2];
        float mx = -3.402823466e38f;
        for (int t = 0; t < T; ++t) mx = fmaxf(mx, sc[t * SLD + s]);
        float sum = 0.0f;
        for (int t = 0; t < T; ++t) sum += __expf(sc[t * SLD + s] - mx);
        const float inv = 1.0f / sum;
        for (int t = 0; t < T; ++t) {
            const float v = sc[t * SLD + s];
            const float r = w0 * fmaxf(v, 0.0f)
                          + w1 * (1.0f / (1.0f + __expf(-v)))
                          + w2 * (__expf(v - mx) * inv);
            act[((long)b * T + t) * T + s] = f32_bf16(r);
        }
    }
}

// ---------------------------------------------------------------------------
// Problem constants:  B=32, T=128, CIN=4096, NF=64, GOUT=4096
// ---------------------------------------------------------------------------
extern "C" void kernel_launch(void* const* d_in, const int* in_sizes, int n_in,
                              void* d_out, int out_size, void* d_ws, size_t ws_size,
                              hipStream_t stream) {
    (void)in_sizes; (void)n_in; (void)out_size; (void)ws_size;

    const float* in1 = (const float*)d_in[0];   // [32,128,64,8,8] -> [4096,4096]
    const float* in2 = (const float*)d_in[1];
    const float* aw  = (const float*)d_in[2];   // [3]
    const float* w1  = (const float*)d_in[3];   // [64,4096]
    const float* b1  = (const float*)d_in[4];
    const float* w2  = (const float*)d_in[5];
    const float* b2  = (const float*)d_in[6];
    const float* wg  = (const float*)d_in[7];   // [4096,4096]
    const float* bg  = (const float*)d_in[8];

    constexpr long M    = 4096;                 // B*T
    constexpr long CIN  = 4096;
    constexpr long NF   = 64;
    constexpr long GOUT = 4096;

    // workspace carve (~100 MB)
    unsigned short* x1b  = (unsigned short*)d_ws;                 // [4096,4096] bf16
    unsigned short* wgb  = x1b + M * CIN;                         // [4096,4096] bf16
    unsigned short* qb   = wgb + GOUT * CIN;                      // [4096, 64]
    unsigned short* kb   = qb + M * NF;                           // [4096, 64]
    unsigned short* gt   = kb + M * NF;                           // [32][4096][128]
    unsigned short* actb = gt + (long)32 * GOUT * 128;            // [32][128][128]

    dim3 blk(256);

    // one-shot bf16 conversion of the L2-resident GEMM operands
    cvt_f32_bf16<<<dim3((unsigned)((M * CIN) / 2048)), blk, 0, stream>>>(in1, x1b, M * CIN);
    cvt_f32_bf16<<<dim3((unsigned)((GOUT * CIN) / 2048)), blk, 0, stream>>>(wg, wgb, GOUT * CIN);

    // q = x1 * w1^T + b1   (f32 in, read once; bf16 out)
    wmma_gemm<float, 64, 1, true><<<dim3(1, 32, 1), blk, 0, stream>>>(
        in1, w1, b1, qb, (int)CIN, (int)NF, 0, 0, 0);
    // k = x2 * w2^T + b2
    wmma_gemm<float, 64, 1, true><<<dim3(1, 32, 1), blk, 0, stream>>>(
        in2, w2, b2, kb, (int)CIN, (int)NF, 0, 0, 0);
    // g = x1 * wg^T + bg   (bf16 in, 128x128 tile; bf16 batch-transposed out)
    wmma_gemm<unsigned short, 128, 2, true><<<dim3(32, 32, 1), blk, 0, stream>>>(
        x1b, wgb, bg, gt, (int)CIN, (int)GOUT, 0, 0, 0);
    // scores + fused activations -> act (bf16)
    scores_act<<<dim3(32), blk, 0, stream>>>(qb, kb, aw, actb);
    // out[b] = act[b] * g_t[b]^T   (f32 out, batched via blockIdx.z)
    wmma_gemm<unsigned short, 128, 0, false><<<dim3(32, 1, 32), blk, 0, stream>>>(
        actb, gt, nullptr, (float*)d_out, 128, (int)GOUT,
        128L * 128L, GOUT * 128L, 128L * GOUT);
}
